// IRadon_4337916969519
// MI455X (gfx1250) — compile-verified, matrix operand order checked by reading, not run on hardware
//
#include <hip/hip_runtime.h>

// ---------------------------------------------------------------------------
// Inverse Radon (filtered backprojection), MI455X / gfx1250, wave32.
//
// Stage 0: sinogram f32 [B,512,180] -> f16 transposed [B,192,512]
//          (angle-major, zero-padded to 192 angles).
// Stage 1: ramp filtering as GEMM  xfilt[b,:,a] = F * x[b,:,a],
//          F[m,k] = 0.5 (m==k), -2/(pi^2 (m-k)^2) ((m-k) odd), else 0.
//          v_wmma_f32_16x16x32_f16. Per K-step:
//            A: 2x ds_load_b128 from a pre-swizzled LDS fragment table
//            B: 2x global_load_b128 (contiguous in transposed f16 layout)
//          Output stored transposed [B,180,512] for stage-2 locality.
// Stage 2: backprojection, 1 thread/pixel, LDS sin/cos tables,
//          contiguous detector-row reads + global_prefetch of next row.
// ---------------------------------------------------------------------------

typedef __attribute__((ext_vector_type(16))) _Float16 v16h;
typedef __attribute__((ext_vector_type(8)))  float    v8f;
typedef __attribute__((ext_vector_type(8)))  unsigned v8u;
typedef __attribute__((ext_vector_type(4)))  unsigned v4u;

#define S_DET   512
#define N_ANG   180
#define N_ANGP  192           // padded to 12 n-tiles of 16
#define N_BATCH 4
#define PI_F    3.14159265358979323846f

// Branchless ramp coefficient: v_rcp_f32 + two v_cndmask, no divergence.
__device__ __forceinline__ _Float16 ramp_h(int d) {
  const float fd = (float)d;
  const float r  = (-2.0f / (PI_F * PI_F)) * __builtin_amdgcn_rcpf(fd * fd);
  float v = (d & 1) ? r : 0.0f;
  v = (d == 0) ? 0.5f : v;
  return (_Float16)v;
}

__device__ __forceinline__ unsigned pack2(_Float16 lo, _Float16 hi) {
  return (unsigned)__builtin_bit_cast(unsigned short, lo) |
         ((unsigned)__builtin_bit_cast(unsigned short, hi) << 16);
}

// ---- Stage 0: f32 [B,512,180] -> f16 [B,192,512] transposed, zero padded ---
__global__ void __launch_bounds__(256)
cvt_transpose(const float* __restrict__ x, _Float16* __restrict__ xhT) {
  const int idx = blockIdx.x * 256 + threadIdx.x;   // B*192*512 elements
  const int k = idx & (S_DET - 1);
  const int a = (idx >> 9) % N_ANGP;
  const int b = idx / (S_DET * N_ANGP);
  const float v = (a < N_ANG) ? x[((size_t)b * S_DET + k) * N_ANG + a] : 0.0f;
  xhT[idx] = (_Float16)v;
}

// ---- Stage 1: one wave per 16x16 tile of xfilt (M=detector, N=angle) ------
__global__ void __launch_bounds__(128)
fbp_filter_wmma(const _Float16* __restrict__ xhT,  // [B,192,512] f16
                float* __restrict__ xf_t) {        // [B,180,512] f32
  // Pre-swizzled A-fragment table: s_F4[u*4+j] = pack(f(d), f(d-1)) with
  // d = (u-511) - 2j.  One K-step chunk (4 packed K-pairs) = one b128 load.
  __shared__ __align__(16) unsigned s_F4[1024 * 4];   // 16 KB
  for (int e = threadIdx.x; e < 1023 * 4; e += 128) {
    const int u = e >> 2;
    const int j = e & 3;
    const int d = (u - 511) - 2 * j;
    s_F4[e] = pack2(ramp_h(d), ramp_h(d - 1));
  }
  __syncthreads();

  const int lane     = threadIdx.x & 31;
  const int wave     = threadIdx.x >> 5;
  const int tile     = blockIdx.x * 4 + wave;      // 1536 tiles
  const int nTiles   = N_ANGP / 16;                // 12
  const int mTiles   = S_DET / 16;                 // 32
  const int b        = tile / (mTiles * nTiles);
  const int rem      = tile - b * (mTiles * nTiles);
  const int mT       = rem / nTiles;
  const int nT       = rem - mT * nTiles;

  const int m0       = mT * 16;
  const int laneHalf = lane >> 4;                  // 0: lanes 0-15, 1: 16-31
  const int mA       = m0 + (lane & 15);           // A row for this lane
  const int nB       = nT * 16 + (lane & 15);      // B column (padded)

  // This lane's B column as packed half2 dwords along K (contiguous).
  const unsigned* __restrict__ xcol =
      (const unsigned*)(xhT + ((size_t)b * N_ANGP + nB) * S_DET);

  // Per-lane constant offsets.
  const int uBase = mA - 8 * laneHalf + 511;       // A table row for kk=0
  const int bOff  = 4 * laneHalf;                  // B dword offset in chunk

  v8f acc = {};
#pragma unroll 4
  for (int kk = 0; kk < S_DET; kk += 32) {
    const int u0 = uBase - kk;                     // chunk v=0..3
    const int u1 = u0 - 16;                        // chunk v=4..7
    const v4u a0 = *(const v4u*)(s_F4 + u0 * 4);   // ds_load_b128
    const v4u a1 = *(const v4u*)(s_F4 + u1 * 4);   // ds_load_b128
    const v4u b0 = *(const v4u*)(xcol + (kk >> 1) + bOff);      // b128
    const v4u b1 = *(const v4u*)(xcol + (kk >> 1) + bOff + 8);  // b128

    const v8u ab = __builtin_shufflevector(a0, a1, 0, 1, 2, 3, 4, 5, 6, 7);
    const v8u bb = __builtin_shufflevector(b0, b1, 0, 1, 2, 3, 4, 5, 6, 7);
    const v16h afrag = __builtin_bit_cast(v16h, ab);
    const v16h bfrag = __builtin_bit_cast(v16h, bb);

    acc = __builtin_amdgcn_wmma_f32_16x16x32_f16(
        /*neg_a=*/false, afrag, /*neg_b=*/false, bfrag,
        /*c_mod=*/(short)0, acc, /*reuse_a=*/false, /*reuse_b=*/false);
  }

  // D layout: lane -> N=lane&15 (halves), VGPR r -> M = r + 8*laneHalf.
  if (nB < N_ANG) {
    float* __restrict__ dst = xf_t + ((size_t)b * N_ANG + nB) * S_DET;
#pragma unroll
    for (int r = 0; r < 8; ++r)
      dst[m0 + r + (laneHalf ? 8 : 0)] = acc[r];
  }
}

// ---- Stage 2: one thread per output pixel, loop over 180 angles -----------
__global__ void __launch_bounds__(256)
fbp_backproject(const float* __restrict__ xf_t,   // [B,180,512]
                float* __restrict__ out) {        // [B,1,512,512]
  __shared__ float s_sin[N_ANG];
  __shared__ float s_cos[N_ANG];

  const int tid = threadIdx.x;
  if (tid < N_ANG) {
    const float th = (float)tid * (PI_F / 180.0f);
    s_sin[tid] = __sinf(th);
    s_cos[tid] = __cosf(th);
  }
  __syncthreads();

  const int idx = blockIdx.x * 256 + tid;          // 4*512*512 threads
  const int j   = idx & (S_DET - 1);               // x (fastest)
  const int i   = (idx >> 9) & (S_DET - 1);        // y
  const int b   = idx >> 18;                       // batch

  const float step = 2.0f / (float)(S_DET - 1);
  const float uj   = -1.0f + step * (float)j;      // xgrid
  const float ui   = -1.0f + step * (float)i;      // ygrid

  const float* __restrict__ base = xf_t + (size_t)b * N_ANG * S_DET;

  float acc = 0.0f;
  for (int a = 0; a < N_ANG; ++a) {
    const float* __restrict__ row = base + (size_t)a * S_DET;
    if (a + 1 < N_ANG)
      __builtin_prefetch(row + S_DET, 0, 0);       // global_prefetch_b8

    const float t  = uj * s_cos[a] - ui * s_sin[a];
    const float p  = fmaf(t, 0.5f * (float)(S_DET - 1),
                             0.5f * (float)(S_DET - 1));
    const float p0 = floorf(p);
    const float w  = p - p0;
    const int   i0 = (int)p0;
    const int   i1 = i0 + 1;
    const float v0 = (i0 >= 0 && i0 < S_DET) ? 1.0f : 0.0f;
    const float v1 = (i1 >= 0 && i1 < S_DET) ? 1.0f : 0.0f;
    const int i0c  = min(max(i0, 0), S_DET - 1);
    const int i1c  = min(max(i1, 0), S_DET - 1);
    acc = fmaf(row[i0c] * v0, 1.0f - w, acc);
    acc = fmaf(row[i1c] * v1, w, acc);
  }

  const float r2   = uj * uj + ui * ui;
  const float norm = PI_F / (2.0f * (float)N_ANG);
  out[idx] = (r2 <= 1.0f) ? acc * norm : 0.0f;
}

extern "C" void kernel_launch(void* const* d_in, const int* in_sizes, int n_in,
                              void* d_out, int out_size, void* d_ws, size_t ws_size,
                              hipStream_t stream) {
  (void)in_sizes; (void)n_in; (void)out_size; (void)ws_size;
  const float* x   = (const float*)d_in[0];
  float*       out = (float*)d_out;

  // Workspace layout (fully rewritten every call):
  //   [0, 786432)          f16 xhT [4,192,512]
  //   [786432, 2260992)    f32 xf_t [4,180,512]
  _Float16* xhT = (_Float16*)d_ws;
  float*    xf  = (float*)((char*)d_ws + (size_t)N_BATCH * N_ANGP * S_DET * 2);

  // Stage 0: 4*192*512 / 256 = 1536 blocks.
  cvt_transpose<<<1536, 256, 0, stream>>>(x, xhT);

  // Stage 1: 1536 WMMA tiles, 4 waves (128 thr) per block -> 384 blocks.
  fbp_filter_wmma<<<384, 128, 0, stream>>>(xhT, xf);

  // Stage 2: one thread per output pixel: 4*512*512 / 256 = 4096 blocks.
  fbp_backproject<<<4096, 256, 0, stream>>>(xf, out);
}
